// LoRABCBlock_32418413150722
// MI455X (gfx1250) — matrix operand order repaired and checked by reading, not compiled, not run on hardware
//
#include <hip/hip_runtime.h>

// ---------------------------------------------------------------------------
// LoRA + attention block for MI455X (gfx1250), wave32, WMMA bf16 path.
// Heavy matmuls: v_wmma_f32_16x16x32_bf16, LDS-staged via
// global_load_async_to_lds_b128 (ASYNCcnt) with double buffering.
// ---------------------------------------------------------------------------

typedef __attribute__((ext_vector_type(16))) __bf16 v16bf;
typedef __attribute__((ext_vector_type(8)))  float  v8f;

#define WMMA_BF16(a, b, c) \
  __builtin_amdgcn_wmma_f32_16x16x32_bf16(false, (a), false, (b), (short)0, (c), false, false)

// ---- Fragment loaders (layouts per CDNA5 ISA §7.12.2) -----------
// A (16x32, bf16, row-major source [ld]): lane L holds row M = L%16,
// K chunks [h*8, h*8+8) and [16+h*8, 16+h*8+8) where h = L/16.
__device__ __forceinline__ v16bf load_frag_a(const __bf16* __restrict__ base,
                                             int ld, int m0, int k0) {
  int lane = threadIdx.x & 31;
  const __bf16* p = base + (size_t)(m0 + (lane & 15)) * ld + k0 + ((lane >> 4) << 3);
  v16bf a;
  ((uint4*)&a)[0] = *(const uint4*)p;        // K = h*8 .. h*8+7
  ((uint4*)&a)[1] = *(const uint4*)(p + 16); // K = 16+h*8 .. 16+h*8+7
  return a;
}

// B (32x16, bf16): source row-major [N][K] (torch Linear weight (out,in)).
// Lane L holds column N = n0 + L%16, K chunks [h*8, h*8+8) and [16+h*8, 16+h*8+8) where h = L/16.
__device__ __forceinline__ v16bf load_frag_b(const __bf16* __restrict__ base,
                                             int ld, int n0, int k0) {
  int lane = threadIdx.x & 31;
  const __bf16* p = base + (size_t)(n0 + (lane & 15)) * ld + k0 + ((lane >> 4) << 3);
  v16bf b;
  ((uint4*)&b)[0] = *(const uint4*)p;
  ((uint4*)&b)[1] = *(const uint4*)(p + 16);
  return b;
}

// ---- CDNA5 async global->LDS copy (tracked by ASYNCcnt) --------------------
__device__ __forceinline__ void async_copy_b128(unsigned lds_addr,
                                                const __bf16* gptr) {
  asm volatile("global_load_async_to_lds_b128 %0, %1, off"
               :: "v"(lds_addr), "v"(gptr) : "memory");
}
__device__ __forceinline__ void wait_asynccnt0() {
  asm volatile("s_wait_asynccnt 0x0" ::: "memory");
}

// ---- Weight conversion f32 -> bf16 -----------------------------------------
__global__ __launch_bounds__(256) void convert_weights_kernel(
    const float* __restrict__ wb, const float* __restrict__ wq,
    const float* __restrict__ wo,
    __bf16* __restrict__ wbb, __bf16* __restrict__ wqb, __bf16* __restrict__ wob) {
  int idx = blockIdx.x * blockDim.x + threadIdx.x;
  if (idx < 1024 * 1024) wbb[idx] = (__bf16)wb[idx];
  if (idx < 192 * 1024)  wqb[idx] = (__bf16)wq[idx];
  if (idx < 1024 * 64)   wob[idx] = (__bf16)wo[idx];
}

// ---- LayerNorm: one block (256 thr) per row of 1024 ------------------------
__global__ __launch_bounds__(256) void layernorm_kernel(
    const float* __restrict__ x, const float* __restrict__ gamma,
    const float* __restrict__ beta,
    float* __restrict__ hn, __bf16* __restrict__ hnb, __bf16* __restrict__ xb) {
  int row = blockIdx.x;
  int tid = threadIdx.x;
  const float* xr = x + (size_t)row * 1024;
  float4 v = ((const float4*)xr)[tid];
  float s  = v.x + v.y + v.z + v.w;
  float ss = v.x * v.x + v.y * v.y + v.z * v.z + v.w * v.w;
  for (int off = 16; off > 0; off >>= 1) {
    s  += __shfl_down(s, off, 32);
    ss += __shfl_down(ss, off, 32);
  }
  __shared__ float ps[8], pss[8];
  __shared__ float mu_s, rs_s;
  int wave = tid >> 5, lane = tid & 31;
  if (lane == 0) { ps[wave] = s; pss[wave] = ss; }
  __syncthreads();
  if (tid == 0) {
    float S = 0.f, SS = 0.f;
    for (int w = 0; w < 8; ++w) { S += ps[w]; SS += pss[w]; }
    float mu = S * (1.0f / 1024.0f);
    float var = SS * (1.0f / 1024.0f) - mu * mu;
    mu_s = mu;
    rs_s = rsqrtf(var + 1e-5f);
  }
  __syncthreads();
  float mu = mu_s, rs = rs_s;
  float4 g  = ((const float4*)gamma)[tid];
  float4 bt = ((const float4*)beta)[tid];
  float4 o;
  o.x = (v.x - mu) * rs * g.x + bt.x;
  o.y = (v.y - mu) * rs * g.y + bt.y;
  o.z = (v.z - mu) * rs * g.z + bt.z;
  o.w = (v.w - mu) * rs * g.w + bt.w;
  ((float4*)(hn + (size_t)row * 1024))[tid] = o;
  size_t cbase = (size_t)row * 1024 + tid * 4;
  hnb[cbase + 0] = (__bf16)o.x; hnb[cbase + 1] = (__bf16)o.y;
  hnb[cbase + 2] = (__bf16)o.z; hnb[cbase + 3] = (__bf16)o.w;
  xb[cbase + 0] = (__bf16)v.x; xb[cbase + 1] = (__bf16)v.y;
  xb[cbase + 2] = (__bf16)v.z; xb[cbase + 3] = (__bf16)v.w;
}

// ---- LDS-staged WMMA GEMM: C[M,N] = A[M,K] * B[N,K]^T (+ Add) -------------
// Block tile 128x128, 8 waves as 2(M)x4(N), each wave 64x32 (4x2 fragments).
// A/B K-slices (128x32 bf16, 8 KB each) staged through LDS with
// global_load_async_to_lds_b128, double-buffered on ASYNCcnt.
// Requires M%128==0, N%128==0, K%32==0.
__global__ __launch_bounds__(256) void wmma_gemm_lds_kernel(
    const __bf16* __restrict__ A, int lda,
    const __bf16* __restrict__ B, int ldb,
    const float* __restrict__ Add, float* __restrict__ C,
    int M, int N, int K) {
  __shared__ __align__(16) __bf16 As[2][128 * 32];
  __shared__ __align__(16) __bf16 Bs[2][128 * 32];
  const int tid  = threadIdx.x;
  const int wave = tid >> 5, lane = tid & 31;
  const int waveM = wave >> 2, waveN = wave & 3;
  const int gm0 = blockIdx.y * 128;
  const int gn0 = blockIdx.x * 128;
  const unsigned bufBytes = 128 * 32 * 2;  // 8 KB per buffer
  const unsigned asBase = (unsigned)(uintptr_t)&As[0][0];
  const unsigned bsBase = (unsigned)(uintptr_t)&Bs[0][0];

  // Stage one 128x32 K-slice of A and B: 512 b128 transfers each tile,
  // 256 threads -> 2 (A) + 2 (B) async issues per thread.
  auto stage = [&](int buf, int k0) {
#pragma unroll
    for (int t = 0; t < 2; ++t) {
      int j = tid + (t << 8);          // 0..511
      int row = j >> 2, seg = j & 3;   // row in tile, 16B segment in 64B row
      const __bf16* ga = A + (size_t)(gm0 + row) * lda + k0 + seg * 8;
      async_copy_b128(asBase + buf * bufBytes + row * 64 + seg * 16, ga);
      const __bf16* gb = B + (size_t)(gn0 + row) * ldb + k0 + seg * 8;
      async_copy_b128(bsBase + buf * bufBytes + row * 64 + seg * 16, gb);
    }
  };

  v8f acc[4][2];
#pragma unroll
  for (int mi = 0; mi < 4; ++mi)
#pragma unroll
    for (int ni = 0; ni < 2; ++ni) { v8f z = {}; acc[mi][ni] = z; }

  stage(0, 0);
  int buf = 0;
  for (int k0 = 0; k0 < K; k0 += 32) {
    wait_asynccnt0();      // our async writes into As/Bs[buf] have landed
    __syncthreads();       // ... and everyone else's too; also: all waves are
                           // done reading the buffer we are about to restage
    if (k0 + 32 < K) stage(buf ^ 1, k0 + 32);

    const __bf16* Ab = &As[buf][0];
    const __bf16* Bb = &Bs[buf][0];
    v16bf af[4], bfr[2];
#pragma unroll
    for (int mi = 0; mi < 4; ++mi)
      af[mi] = load_frag_a(Ab, 32, waveM * 64 + mi * 16, 0);
#pragma unroll
    for (int ni = 0; ni < 2; ++ni)
      bfr[ni] = load_frag_b(Bb, 32, waveN * 32 + ni * 16, 0);
#pragma unroll
    for (int mi = 0; mi < 4; ++mi)
#pragma unroll
      for (int ni = 0; ni < 2; ++ni)
        acc[mi][ni] = WMMA_BF16(af[mi], bfr[ni], acc[mi][ni]);
    buf ^= 1;
  }

  int rb = (lane >> 4) << 3, cc = lane & 15;
#pragma unroll
  for (int mi = 0; mi < 4; ++mi) {
    int mbase = gm0 + waveM * 64 + mi * 16;
#pragma unroll
    for (int ni = 0; ni < 2; ++ni) {
      int nbase = gn0 + waveN * 32 + ni * 16;
#pragma unroll
      for (int i = 0; i < 8; ++i) {
        size_t idx = (size_t)(mbase + rb + i) * N + nbase + cc;
        float v = acc[mi][ni][i];
        if (Add) v += Add[idx];
        C[idx] = v;
      }
    }
  }
}

// ---- Direct WMMA GEMM (generic shapes, used for N=192 qkv projection) ------
__global__ __launch_bounds__(256) void wmma_gemm_nt_kernel(
    const __bf16* __restrict__ A, int lda,
    const __bf16* __restrict__ B, int ldb,
    const float* __restrict__ Add, float* __restrict__ C,
    int M, int N, int K) {
  int wave = threadIdx.x >> 5;
  int lane = threadIdx.x & 31;
  int tilesN = N >> 5;
  int tilesM = M >> 5;
  long tile = (long)blockIdx.x * 8 + wave;
  if (tile >= (long)tilesM * tilesN) return;  // wave-uniform exit
  int m0 = (int)(tile / tilesN) << 5;
  int n0 = (int)(tile % tilesN) << 5;

  v8f c00 = {}, c01 = {}, c10 = {}, c11 = {};
  for (int k0 = 0; k0 < K; k0 += 32) {
    v16bf a0 = load_frag_a(A, lda, m0,      k0);
    v16bf a1 = load_frag_a(A, lda, m0 + 16, k0);
    v16bf b0 = load_frag_b(B, ldb, n0,      k0);
    v16bf b1 = load_frag_b(B, ldb, n0 + 16, k0);
    c00 = WMMA_BF16(a0, b0, c00);
    c01 = WMMA_BF16(a0, b1, c01);
    c10 = WMMA_BF16(a1, b0, c10);
    c11 = WMMA_BF16(a1, b1, c11);
    if (k0 + 64 < K) {  // global_prefetch_b8
      __builtin_prefetch(A + (size_t)(m0 + (lane & 15)) * lda + k0 + 64, 0, 1);
      __builtin_prefetch(B + (size_t)(n0 + (lane & 15)) * ldb + k0 + 64, 0, 1);
    }
  }

  int rb = (lane >> 4) << 3, cc = lane & 15;
#pragma unroll
  for (int i = 0; i < 8; ++i) {
    size_t r0 = (size_t)(m0 + rb + i) * N;
    size_t r1 = (size_t)(m0 + 16 + rb + i) * N;
    float o00 = c00[i], o01 = c01[i], o10 = c10[i], o11 = c11[i];
    if (Add) {
      o00 += Add[r0 + n0 + cc];      o01 += Add[r0 + n0 + 16 + cc];
      o10 += Add[r1 + n0 + cc];      o11 += Add[r1 + n0 + 16 + cc];
    }
    C[r0 + n0 + cc] = o00;  C[r0 + n0 + 16 + cc] = o01;
    C[r1 + n0 + cc] = o10;  C[r1 + n0 + 16 + cc] = o11;
  }
}

// ---- QKV repack: split heads, scale q, transpose v -------------------------
__global__ __launch_bounds__(256) void qkv_pack_kernel(
    const float* __restrict__ qkv, __bf16* __restrict__ qb,
    __bf16* __restrict__ kb, __bf16* __restrict__ vt, int S) {
  int idx = blockIdx.x * blockDim.x + threadIdx.x;   // 8192*64 threads
  int row = idx >> 6;
  int c = idx & 63;
  int h = c >> 5, d = c & 31;
  int b = row / S, s = row % S;
  size_t head = (size_t)(b * 2 + h);
  const float* src = qkv + (size_t)row * 192;
  const float scale = 0.17677669529663687f;  // 1/sqrt(32)
  qb[(head * S + s) * 32 + d] = (__bf16)(src[h * 32 + d] * scale);
  kb[(head * S + s) * 32 + d] = (__bf16)(src[64 + h * 32 + d]);
  vt[(head * 32 + d) * S + s] = (__bf16)(src[128 + h * 32 + d]);
}

// ---- Flash attention: 8 waves/block, 16 q-rows per wave, stream keys -------
__global__ __launch_bounds__(256) void flash_attn_kernel(
    const __bf16* __restrict__ qb, const __bf16* __restrict__ kb,
    const __bf16* __restrict__ vt, __bf16* __restrict__ attn_out, int S) {
  const int HD = 32;
  int h = blockIdx.y, b = blockIdx.z;
  int wave = threadIdx.x >> 5, lane = threadIdx.x & 31;
  int q0 = (blockIdx.x * 8 + wave) * 16;
  size_t head = (size_t)(b * 2 + h);
  const __bf16* Q  = qb + head * S * HD;
  const __bf16* Km = kb + head * S * HD;
  const __bf16* Vt = vt + head * HD * S;
  __shared__ __align__(16) __bf16 pbuf[8][16 * 32];  // per-wave P (C->A layout)

  v16bf aq = load_frag_a(Q, HD, q0, 0);
  v8f o0 = {}, o1 = {};
  float mrow[8], lrow[8];
#pragma unroll
  for (int i = 0; i < 8; ++i) { mrow[i] = -1e30f; lrow[i] = 0.f; }

  int rb = (lane >> 4) << 3, cc = lane & 15;
  for (int kc = 0; kc < S; kc += 32) {
    v16bf bk0 = load_frag_b(Km, HD, kc,      0);
    v16bf bk1 = load_frag_b(Km, HD, kc + 16, 0);
    v8f s0 = {}, s1 = {};
    s0 = WMMA_BF16(aq, bk0, s0);
    s1 = WMMA_BF16(aq, bk1, s1);

    float nm[8], al[8];
#pragma unroll
    for (int i = 0; i < 8; ++i) {
      float tm = fmaxf(s0[i], s1[i]);
      tm = fmaxf(tm, __shfl_xor(tm, 1, 32));
      tm = fmaxf(tm, __shfl_xor(tm, 2, 32));
      tm = fmaxf(tm, __shfl_xor(tm, 4, 32));
      tm = fmaxf(tm, __shfl_xor(tm, 8, 32));  // stays within 16-lane half
      nm[i] = fmaxf(mrow[i], tm);
      al[i] = __expf(mrow[i] - nm[i]);
      mrow[i] = nm[i];
    }
#pragma unroll
    for (int i = 0; i < 8; ++i) {
      float p0 = __expf(s0[i] - nm[i]);
      float p1 = __expf(s1[i] - nm[i]);
      pbuf[wave][(rb + i) * 32 + cc]      = (__bf16)p0;
      pbuf[wave][(rb + i) * 32 + 16 + cc] = (__bf16)p1;
      float rs = p0 + p1;
      rs += __shfl_xor(rs, 1, 32);
      rs += __shfl_xor(rs, 2, 32);
      rs += __shfl_xor(rs, 4, 32);
      rs += __shfl_xor(rs, 8, 32);
      lrow[i] = lrow[i] * al[i] + rs;
      o0[i] *= al[i];
      o1[i] *= al[i];
    }
    v16bf ap  = load_frag_a(&pbuf[wave][0], 32, 0, 0);
    v16bf bv0 = load_frag_b(Vt, S, 0,  kc);
    v16bf bv1 = load_frag_b(Vt, S, 16, kc);
    o0 = WMMA_BF16(ap, bv0, o0);
    o1 = WMMA_BF16(ap, bv1, o1);
  }

  size_t row0 = (size_t)b * S + q0;
#pragma unroll
  for (int i = 0; i < 8; ++i) {
    float inv = 1.0f / lrow[i];
    size_t r = (row0 + rb + i) * 64 + h * 32;
    attn_out[r + cc]      = (__bf16)(o0[i] * inv);
    attn_out[r + 16 + cc] = (__bf16)(o1[i] * inv);
  }
}

// ---- LoRA down: t[m, 0..7] = h_with_attn[m, :] @ lora_down -----------------
__global__ __launch_bounds__(256) void lora_down_kernel(
    const float* __restrict__ h, const float* __restrict__ down,
    float* __restrict__ t) {
  int m = blockIdx.x;
  const float* row = h + (size_t)m * 1024;
  float acc[8] = {0.f, 0.f, 0.f, 0.f, 0.f, 0.f, 0.f, 0.f};
  for (int k = threadIdx.x; k < 1024; k += blockDim.x) {
    float xv = row[k];
    const float* dk = down + k * 8;
#pragma unroll
    for (int r = 0; r < 8; ++r) acc[r] += xv * dk[r];
  }
#pragma unroll
  for (int r = 0; r < 8; ++r)
    for (int off = 16; off > 0; off >>= 1)
      acc[r] += __shfl_down(acc[r], off, 32);
  __shared__ float part[8][8];
  int wave = threadIdx.x >> 5, lane = threadIdx.x & 31;
  if (lane == 0)
    for (int r = 0; r < 8; ++r) part[wave][r] = acc[r];
  __syncthreads();
  if (threadIdx.x < 8) {
    float s = 0.f;
    for (int w = 0; w < 8; ++w) s += part[w][threadIdx.x];
    t[(size_t)m * 8 + threadIdx.x] = s;
  }
}

// ---- Final: out += SCALING * (t @ lora_up) ---------------------------------
__global__ __launch_bounds__(256) void lora_up_add_kernel(
    const float* __restrict__ t, const float* __restrict__ up,
    float* __restrict__ out, float scaling) {
  size_t idx = (size_t)blockIdx.x * blockDim.x + threadIdx.x;
  int m = (int)(idx >> 10);
  int n = (int)(idx & 1023);
  const float* tm = t + (size_t)m * 8;
  float acc = 0.f;
#pragma unroll
  for (int r = 0; r < 8; ++r) acc += tm[r] * up[r * 1024 + n];
  out[idx] += scaling * acc;
}

// ---------------------------------------------------------------------------
extern "C" void kernel_launch(void* const* d_in, const int* in_sizes, int n_in,
                              void* d_out, int out_size, void* d_ws, size_t ws_size,
                              hipStream_t stream) {
  const int B = 4, S = 2048, E = 1024, M = B * S;  // 8192 rows
  const float* x       = (const float*)d_in[0];
  const float* w_base  = (const float*)d_in[1];
  const float* ln_g    = (const float*)d_in[2];
  const float* ln_b    = (const float*)d_in[3];
  const float* ld_w    = (const float*)d_in[4];  // lora_down [1024][8]
  const float* lu_w    = (const float*)d_in[5];  // lora_up   [8][1024]
  const float* w_qkv   = (const float*)d_in[6];  // [192][1024]
  const float* w_ao    = (const float*)d_in[7];  // [1024][64]
  float* out = (float*)d_out;

  char* ws = (char*)d_ws;
  auto alloc = [&](size_t bytes) {
    char* p = ws;
    ws += (bytes + 255) & ~(size_t)255;
    return p;
  };
  __bf16* x_bf   = (__bf16*)alloc((size_t)M * E * 2);
  __bf16* h_bf   = (__bf16*)alloc((size_t)M * E * 2);
  float*  h_f32  = (float*) alloc((size_t)M * E * 4);
  __bf16* wb_bf  = (__bf16*)alloc((size_t)E * E * 2);
  __bf16* wq_bf  = (__bf16*)alloc((size_t)192 * E * 2);
  __bf16* wo_bf  = (__bf16*)alloc((size_t)E * 64 * 2);
  float*  qkv    = (float*) alloc((size_t)M * 192 * 4);
  __bf16* q_bf   = (__bf16*)alloc((size_t)M * 32 * 2 * 2);
  __bf16* k_bf   = (__bf16*)alloc((size_t)M * 32 * 2 * 2);
  __bf16* vt_bf  = (__bf16*)alloc((size_t)M * 32 * 2 * 2);
  __bf16* attn   = (__bf16*)alloc((size_t)M * 64 * 2);
  float*  hattn  = (float*) alloc((size_t)M * E * 4);
  float*  t_lo   = (float*) alloc((size_t)M * 8 * 4);

  // 1) f32 -> bf16 weights
  convert_weights_kernel<<<(E * E) / 256, 256, 0, stream>>>(
      w_base, w_qkv, w_ao, wb_bf, wq_bf, wo_bf);

  // 2) LayerNorm (also emits bf16 copies of x and h_norm)
  layernorm_kernel<<<M, 256, 0, stream>>>(x, ln_g, ln_b, h_f32, h_bf, x_bf);

  // 3) base_output = x @ w_base^T  (LDS/async staged, writes d_out)
  wmma_gemm_lds_kernel<<<dim3(E / 128, M / 128), 256, 0, stream>>>(
      x_bf, E, wb_bf, E, nullptr, out, M, E, E);

  // 4) qkv = h_norm @ w_qkv^T  (N = 192, direct kernel)
  {
    int tiles = (M / 32) * (192 / 32);
    wmma_gemm_nt_kernel<<<(tiles + 7) / 8, 256, 0, stream>>>(
        h_bf, E, wq_bf, E, nullptr, qkv, M, 192, E);
  }
  // 5) split/scale/transpose heads
  qkv_pack_kernel<<<(M * 64) / 256, 256, 0, stream>>>(qkv, q_bf, k_bf, vt_bf, S);

  // 6) flash attention
  flash_attn_kernel<<<dim3(S / 128, 2, B), 256, 0, stream>>>(q_bf, k_bf, vt_bf, attn, S);

  // 7) h_with_attn = h_norm + attn @ w_attn_out^T  (LDS/async staged)
  wmma_gemm_lds_kernel<<<dim3(E / 128, M / 128), 256, 0, stream>>>(
      attn, 64, wo_bf, 64, h_f32, hattn, M, E, 64);

  // 8) rank-8 LoRA down
  lora_down_kernel<<<M, 256, 0, stream>>>(hattn, ld_w, t_lo);

  // 9) out += SCALING * (t @ lora_up)   (SCALING = 1/8)
  lora_up_add_kernel<<<(M * E) / 256, 256, 0, stream>>>(t_lo, lu_w, out, 0.125f);

  (void)in_sizes; (void)n_in; (void)out_size; (void)ws_size;
}
